// GCNmodel_70351564308950
// MI455X (gfx1250) — compile-verified, matrix operand order checked by reading, not run on hardware
//
#include <hip/hip_runtime.h>
#include <hip/hip_bf16.h>

typedef __attribute__((ext_vector_type(16))) __bf16   v16bf;
typedef __attribute__((ext_vector_type(8)))  float    v8f;
typedef __attribute__((ext_vector_type(8)))  unsigned v8u;

#define NFEAT 128
#define LDSB  136   // LDS row stride in bf16 elements (272B = 16*17 -> conflict-free, 16B aligned)

// ---------- helpers ----------
__device__ __forceinline__ unsigned f2bf(float f) {
    unsigned u = __builtin_bit_cast(unsigned, f);
    unsigned r = u + 0x7FFFu + ((u >> 16) & 1u);   // round-to-nearest-even
    return r >> 16;
}
__device__ __forceinline__ unsigned pack2bf(float lo, float hi) {
    return f2bf(lo) | (f2bf(hi) << 16);
}
__device__ __forceinline__ unsigned hash_u32(unsigned x) {
    x ^= x >> 16; x *= 0x7feb352du;
    x ^= x >> 15; x *= 0x846ca68bu;
    x ^= x >> 16; return x;
}

// ---------- small utility kernels ----------
__global__ void fill1_kernel(float* p, int n) {
    int i = blockIdx.x * blockDim.x + threadIdx.x;
    if (i < n) p[i] = 1.0f;   // self-loop contribution to degree
}

__global__ void degree_kernel(const long long* __restrict__ dst, float* deg, int E) {
    int e = blockIdx.x * blockDim.x + threadIdx.x;
    if (e < E) atomicAdd(&deg[(int)dst[e]], 1.0f);
}

__global__ void rsqrt_kernel(float* p, int n) {
    int i = blockIdx.x * blockDim.x + threadIdx.x;
    if (i < n) p[i] = rsqrtf(p[i]);   // deg >= 1 always
}

// Pack W (K=128 x N=128, row-major) into WMMA bf16 B-fragment order with
// per-lane-contiguous dwords: index = (((nt*4 + ks)*32 + lane)*8 + v)
__global__ void pack_w_kernel(const float* __restrict__ W, unsigned* __restrict__ Wp) {
    int tid = blockIdx.x * blockDim.x + threadIdx.x;
    if (tid >= 8 * 4 * 32 * 8) return;
    int v    = tid & 7;
    int lane = (tid >> 3) & 31;
    int ks   = (tid >> 8) & 3;
    int nt   = tid >> 10;
    int n    = nt * 16 + (lane & 15);
    int kg   = lane >> 4;
    int k    = ks * 32 + ((v < 4) ? 0 : 16) + kg * 8 + (v & 3) * 2;
    Wp[tid]  = pack2bf(W[k * NFEAT + n], W[(k + 1) * NFEAT + n]);
}

// ---------- GEMM: H[M,128] = X[M,128] @ W[128,128] via v_wmma_f32_16x16x32_bf16 ----------
// Block = 256 threads = 8 waves. Block computes a 16-row x 128-col slab.
// Wave w computes the 16x16 tile at columns [16w, 16w+16).
// A is staged in LDS as bf16 so fragment loads are two ds_load_b128 per K-step.
__global__ void gemm_bf16_kernel(const float* __restrict__ X,
                                 const unsigned* __restrict__ Wp,
                                 float* __restrict__ H, int nrows) {
    __shared__ unsigned short AsB[16 * LDSB];

    const int t    = threadIdx.x;
    const int wave = t >> 5;
    const int lane = t & 31;
    const int row0 = blockIdx.x * 16;
    const bool full = (row0 + 16 <= nrows);   // fast path: whole tile in range

    // Stage A tile (16 x 128), converting f32 -> bf16 once.
    // Each thread: 8 contiguous f32 -> 4 packed dwords -> one 16B LDS store.
    {
        int r  = t >> 4;          // 0..15
        int cb = (t & 15) * 8;    // 0,8,...,120
        unsigned p0 = 0, p1 = 0, p2 = 0, p3 = 0;
        if (full || (row0 + r < nrows)) {
            const float4* p = (const float4*)(X + (long long)(row0 + r) * NFEAT + cb);
            float4 v0 = p[0], v1 = p[1];
            p0 = pack2bf(v0.x, v0.y);
            p1 = pack2bf(v0.z, v0.w);
            p2 = pack2bf(v1.x, v1.y);
            p3 = pack2bf(v1.z, v1.w);
        }
        *(uint4*)(&AsB[r * LDSB + cb]) = make_uint4(p0, p1, p2, p3);
    }
    __syncthreads();

    const int m  = lane & 15;
    const int kg = lane >> 4;
    const unsigned short* arow = &AsB[m * LDSB + kg * 8];

    v8f c = {};
    #pragma unroll
    for (int ks = 0; ks < 4; ++ks) {
        // A fragment: v0..3 = elements [ks*32 + kg*8, +8), v4..7 = +16 further
        uint4 a0 = *(const uint4*)(arow + ks * 32);
        uint4 a1 = *(const uint4*)(arow + ks * 32 + 16);
        // B fragment: 8 contiguous dwords per lane
        const uint4* bp = (const uint4*)(Wp + ((((wave * 4 + ks) * 32) + lane) << 3));
        uint4 b0 = bp[0], b1 = bp[1];

        v8u apk = {a0.x, a0.y, a0.z, a0.w, a1.x, a1.y, a1.z, a1.w};
        v8u bpk = {b0.x, b0.y, b0.z, b0.w, b1.x, b1.y, b1.z, b1.w};
        v16bf a = __builtin_bit_cast(v16bf, apk);
        v16bf b = __builtin_bit_cast(v16bf, bpk);
        c = __builtin_amdgcn_wmma_f32_16x16x32_bf16(
                /*neg_a=*/false, a, /*neg_b=*/false, b,
                /*c_mod=*/(short)0, c, /*reuse_a=*/false, /*reuse_b=*/false);
    }

    // C/D layout: VGPR r -> M = r + (lane>=16)*8 ; N = lane&15
    const int n     = wave * 16 + (lane & 15);
    const int mbase = (lane >> 4) * 8;
    float* hp = H + (long long)(row0 + mbase) * NFEAT + n;
    if (full) {
        #pragma unroll
        for (int r = 0; r < 8; ++r) {
            hp[(long long)r * NFEAT] = c[r];
        }
    } else {
        #pragma unroll
        for (int r = 0; r < 8; ++r) {
            if (row0 + mbase + r < nrows) hp[(long long)r * NFEAT] = c[r];
        }
    }
}

// out[i,f] = bias[f] + h[i,f] * dinv[i]^2   (self-loop term + bias seed)
__global__ void init_self_kernel(const float* __restrict__ h,
                                 const float* __restrict__ dinv,
                                 const float* __restrict__ bias,
                                 float* __restrict__ out, int n_nodes) {
    long long idx = (long long)blockIdx.x * blockDim.x + threadIdx.x;
    if (idx >= (long long)n_nodes * NFEAT) return;
    int i = (int)(idx >> 7);
    int f = (int)(idx & 127);
    float d = dinv[i];
    out[idx] = bias[f] + h[idx] * d * d;
}

// 32 lanes per edge, float4 per lane: out[dst] += h[src] * dinv[src]*dinv[dst]
__global__ void aggregate_kernel(const long long* __restrict__ src,
                                 const long long* __restrict__ dst,
                                 const float* __restrict__ dinv,
                                 const float* __restrict__ h,
                                 float* __restrict__ out, int E) {
    long long gid = (long long)blockIdx.x * blockDim.x + threadIdx.x;
    int e = (int)(gid >> 5);
    int l = (int)(gid & 31);
    if (e >= E) return;
    int s = (int)src[e];
    int d = (int)dst[e];
    float w = dinv[s] * dinv[d];
    const float4 v = *(const float4*)(h + (long long)s * NFEAT + l * 4);
    float* o = out + (long long)d * NFEAT + l * 4;
    atomicAdd(o + 0, v.x * w);
    atomicAdd(o + 1, v.y * w);
    atomicAdd(o + 2, v.z * w);
    atomicAdd(o + 3, v.w * w);
}

// a = dropout(relu(x), p=0.2) with deterministic hash mask, inverted scaling
__global__ void relu_dropout_kernel(const float* __restrict__ in,
                                    float* __restrict__ out, long long n) {
    long long idx = (long long)blockIdx.x * blockDim.x + threadIdx.x;
    if (idx >= n) return;
    float v = in[idx];
    v = v > 0.0f ? v : 0.0f;
    unsigned hsh = hash_u32((unsigned)idx ^ 0x2A2A2A2Au);
    bool keep = (hsh & 0xFFFFFFu) < 13421773u;   // 0.8 * 2^24
    out[idx] = keep ? v * 1.25f : 0.0f;
}

// ---------- launch ----------
extern "C" void kernel_launch(void* const* d_in, const int* in_sizes, int n_in,
                              void* d_out, int out_size, void* d_ws, size_t ws_size,
                              hipStream_t stream) {
    const float*     x  = (const float*)d_in[0];
    const long long* ei = (const long long*)d_in[1];   // [2, E] int64
    const float*     W1 = (const float*)d_in[2];
    const float*     b1 = (const float*)d_in[3];
    const float*     W2 = (const float*)d_in[4];
    const float*     b2 = (const float*)d_in[5];
    float*           out = (float*)d_out;

    const int N = in_sizes[0] / NFEAT;
    const int E = in_sizes[1] / 2;
    const long long NF = (long long)N * NFEAT;

    // workspace carve-up (512B aligned)
    char* ws = (char*)d_ws;
    size_t off = 0;
    auto carve = [&](size_t bytes) -> void* {
        void* p = ws + off;
        off = (off + bytes + 511) & ~(size_t)511;
        return p;
    };
    float*    dinv = (float*)carve((size_t)N * 4);
    unsigned* W1p  = (unsigned*)carve(8192 * 4);
    unsigned* W2p  = (unsigned*)carve(8192 * 4);
    float*    bufA = (float*)carve((size_t)NF * 4);
    float*    bufB = (float*)carve((size_t)NF * 4);

    const int TB = 256;
    const int nBlkN   = (N + TB - 1) / TB;
    const int nBlkE   = (E + TB - 1) / TB;
    const int nBlkNF  = (int)((NF + TB - 1) / TB);
    const int nBlkAgg = (int)(((long long)E * 32 + TB - 1) / TB);
    const int nBlkGemm = (N + 15) / 16;

    // degree -> dinv
    fill1_kernel<<<nBlkN, TB, 0, stream>>>(dinv, N);
    degree_kernel<<<nBlkE, TB, 0, stream>>>(ei + E, dinv, E);
    rsqrt_kernel<<<nBlkN, TB, 0, stream>>>(dinv, N);

    // pack weights to WMMA bf16 fragments
    pack_w_kernel<<<32, TB, 0, stream>>>(W1, W1p);
    pack_w_kernel<<<32, TB, 0, stream>>>(W2, W2p);

    // layer 1: h1 = x@W1 ; agg1 = scatter(norm * h1) + b1
    gemm_bf16_kernel<<<nBlkGemm, TB, 0, stream>>>(x, W1p, bufA, N);
    init_self_kernel<<<nBlkNF, TB, 0, stream>>>(bufA, dinv, b1, bufB, N);
    aggregate_kernel<<<nBlkAgg, TB, 0, stream>>>(ei, ei + E, dinv, bufA, bufB, E);

    // relu + dropout -> bufA
    relu_dropout_kernel<<<nBlkNF, TB, 0, stream>>>(bufB, bufA, NF);

    // layer 2: h2 = a@W2 ; out = scatter(norm * h2) + b2
    gemm_bf16_kernel<<<nBlkGemm, TB, 0, stream>>>(bufA, W2p, bufB, N);
    init_self_kernel<<<nBlkNF, TB, 0, stream>>>(bufB, dinv, b2, out, N);
    aggregate_kernel<<<nBlkAgg, TB, 0, stream>>>(ei, ei + E, dinv, bufB, out, E);
}
